// PreT_Attention_52106543235483
// MI455X (gfx1250) — compile-verified, matrix operand order checked by reading, not run on hardware
//
#include <hip/hip_runtime.h>
#include <cstdint>

typedef __attribute__((ext_vector_type(16))) __bf16 v16bf;
typedef __attribute__((ext_vector_type(8)))  float  v8f;

#define NH    12
#define B_    64
#define N_    197
#define C_    768
#define D_    64
#define P_    10
#define MPAD  224              // (P+N)=207 padded to 7*32
#define F_    (3*C_)           // 2304
#define MROWS (B_*N_)          // 12608
#define NKSTEP (C_/32)         // 24 k-steps

// ---------- helpers ----------
__device__ __forceinline__ uint16_t f2bf(float f) {
  uint32_t x = __float_as_uint(f);
  uint32_t r = x + 0x7FFFu + ((x >> 16) & 1u);   // round-to-nearest-even
  return (uint16_t)(r >> 16);
}

union FragU { v16bf v; uint4 q[2]; };

// Per-lane fragment for 16-bit WMMA A/B layout:
// lane<16 holds K=[kb..kb+7] and [kb+16..kb+23]; two 16-B chunks 32 B apart.
__device__ __forceinline__ v16bf make_frag(const uint16_t* p) {
  FragU u;
  const uint4* q = (const uint4*)p;
  u.q[0] = q[0];
  u.q[1] = q[2];
  return u.v;
}

__device__ __forceinline__ v8f wmma_bf16(v16bf a, v16bf b, v8f c) {
  return __builtin_amdgcn_wmma_f32_16x16x32_bf16(false, a, false, b, (short)0, c,
                                                 false, false);
}

// ---- CDNA5 async global->LDS DMA (ASYNCcnt-tracked), 16B per lane ----
__device__ __forceinline__ void async_cp16(const uint16_t* lds_ptr,
                                           const uint16_t* gptr) {
  uint32_t lds_off = (uint32_t)(uintptr_t)lds_ptr;   // low 32 bits = LDS offset
  asm volatile("global_load_async_to_lds_b128 %0, %1, off"
               :: "v"(lds_off), "v"(gptr) : "memory");
}

template <int N>
__device__ __forceinline__ void wait_async() {
  asm volatile("s_wait_asynccnt %0" :: "i"(N) : "memory");
}

// Cooperative stage of ROWSx32 bf16 tile by 128 threads, b128 per chunk.
template <int ROWS>
__device__ __forceinline__ void stage_tile(const uint16_t* gbase, int k,
                                           uint16_t (*lbuf)[32], int tid) {
  #pragma unroll
  for (int s = 0; s < ROWS / 32; ++s) {   // ROWS*4 chunks / 128 threads
    int c   = tid + s * 128;
    int row = c >> 2;
    int ci  = (c & 3) * 8;
    async_cp16(&lbuf[row][ci], gbase + (size_t)row * C_ + k + ci);
  }
}

// 8 WMMAs of one k-step: per-wave 32Mx64N from staged LDS tiles.
__device__ __forceinline__ void gemm_step(const uint16_t (*a_s)[32],
                                          const uint16_t (*b_s)[32],
                                          int am0, int bn0,
                                          int llow, int kboff, v8f acc[8]) {
  v16bf a0 = make_frag(&a_s[am0 +      llow][kboff]);
  v16bf a1 = make_frag(&a_s[am0 + 16 + llow][kboff]);
  #pragma unroll
  for (int t = 0; t < 4; ++t) {
    v16bf bf = make_frag(&b_s[bn0 + t*16 + llow][kboff]);
    acc[t]     = wmma_bf16(a0, bf, acc[t]);
    acc[4 + t] = wmma_bf16(a1, bf, acc[4 + t]);
  }
}

// Shared double-buffered async mainloop: A 64x768, B 128x768 slabs.
__device__ __forceinline__ void gemm_mainloop(const uint16_t* Ab, const uint16_t* Bb,
                                              uint16_t (*a_s)[64][32],
                                              uint16_t (*b_s)[128][32],
                                              int tid, int wm, int wn,
                                              int llow, int kboff, v8f acc[8]) {
  stage_tile<64 >(Ab, 0, a_s[0], tid);
  stage_tile<128>(Bb, 0, b_s[0], tid);
  for (int i = 0; i < NKSTEP - 1; ++i) {
    const int cur = i & 1, nxt = cur ^ 1;
    stage_tile<64 >(Ab, (i + 1) * 32, a_s[nxt], tid);
    stage_tile<128>(Bb, (i + 1) * 32, b_s[nxt], tid);
    wait_async<6>();                 // stage i landed; stage i+1 in flight
    __syncthreads();
    gemm_step(a_s[cur], b_s[cur], wm * 32, wn * 64, llow, kboff, acc);
    __syncthreads();                 // buffer free before refill
  }
  wait_async<0>();
  __syncthreads();
  gemm_step(a_s[(NKSTEP - 1) & 1], b_s[(NKSTEP - 1) & 1],
            wm * 32, wn * 64, llow, kboff, acc);
}

// ---------- fp32 -> bf16 conversion ----------
__global__ void k_cvt(const float* __restrict__ src, uint16_t* __restrict__ dst, int n) {
  int i = blockIdx.x * blockDim.x + threadIdx.x;
  int stride = gridDim.x * blockDim.x;
  for (; i < n; i += stride) dst[i] = f2bf(src[i]);
}

// ---------- prompt prefix into K / V^T + zero padding ----------
__global__ void k_prefix(const float* __restrict__ prompt,
                         uint16_t* __restrict__ kb, uint16_t* __restrict__ vt) {
  int bh = blockIdx.x;
  int b = bh / NH, h = bh % NH;
  int d = threadIdx.x;                 // 0..63
  for (int p = 0; p < P_; ++p) {
    float kv = prompt[(((size_t)(b*2 + 0)*P_ + p)*NH + h)*D_ + d];
    float vv = prompt[(((size_t)(b*2 + 1)*P_ + p)*NH + h)*D_ + d];
    kb[((size_t)bh*MPAD + p)*D_ + d]  = f2bf(kv);
    vt[((size_t)bh*D_ + d)*MPAD + p]  = f2bf(vv);
  }
  for (int m = P_ + N_; m < MPAD; ++m) {
    kb[((size_t)bh*MPAD + m)*D_ + d] = 0;
    vt[((size_t)bh*D_ + d)*MPAD + m] = 0;
  }
}

// ---------- QKV GEMM: 64Mx128N workgroup tile (2x2 waves of 32Mx64N) ----------
__global__ void __launch_bounds__(128)
k_gemm_qkv(const uint16_t* __restrict__ A,   // x bf16 [MROWS x C]
           const uint16_t* __restrict__ Bt,  // qkv_w bf16 [F x C]
           uint16_t* __restrict__ qb,        // [BH x N x D]
           uint16_t* __restrict__ kb,        // [BH x MPAD x D]
           uint16_t* __restrict__ vt) {      // [BH x D x MPAD]
  __shared__ __align__(16) uint16_t a_s[2][64][32];
  __shared__ __align__(16) uint16_t b_s[2][128][32];
  const int tid   = threadIdx.x;
  const int wave  = tid >> 5;
  const int wm    = wave & 1, wn = wave >> 1;
  const int lane  = tid & 31;
  const int llow  = lane & 15;
  const int kboff = (lane >> 4) << 3;
  const int m0 = blockIdx.x * 64;
  const int n0 = blockIdx.y * 128;

  v8f acc[8];
  #pragma unroll
  for (int i = 0; i < 8; ++i) acc[i] = (v8f){};

  gemm_mainloop(A + (size_t)m0 * C_, Bt + (size_t)n0 * C_,
                a_s, b_s, tid, wm, wn, llow, kboff, acc);

  #pragma unroll
  for (int r = 0; r < 2; ++r) {
    #pragma unroll
    for (int t = 0; t < 4; ++t) {
      int col   = n0 + wn*64 + t*16 + llow;  // 0..2303
      int which = col / C_;                  // 0=q 1=k 2=v
      int rem   = col - which*C_;
      int h = rem >> 6, d = rem & 63;
      #pragma unroll
      for (int j = 0; j < 8; ++j) {
        int row = m0 + wm*32 + r*16 + j + kboff;
        int b = row / N_;
        int n = row - b*N_;
        int bh = b*NH + h;
        uint16_t bv = f2bf(acc[r*4 + t][j]);
        if (which == 0)      qb[((size_t)bh*N_   + n)      *D_   + d]      = bv;
        else if (which == 1) kb[((size_t)bh*MPAD + (P_+n)) *D_   + d]      = bv;
        else                 vt[((size_t)bh*D_   + d)      *MPAD + (P_+n)] = bv;
      }
    }
  }
}

// ---------- attention: per (b,h,16-query-tile) wave ----------
__global__ void k_attn(const uint16_t* __restrict__ qb, const uint16_t* __restrict__ kb,
                       const uint16_t* __restrict__ vt, uint16_t* __restrict__ ao) {
  __shared__ __align__(16) float    s_sc[16][MPAD];
  __shared__ __align__(16) uint16_t s_pr[16][MPAD];
  const int lane  = threadIdx.x & 31;
  const int llow  = lane & 15;
  const int kboff = (lane >> 4) << 3;
  const int n0 = blockIdx.x * 16;
  const int h  = blockIdx.y, b = blockIdx.z;
  const int bh = b*NH + h;
  const uint16_t* qbh = qb + (size_t)bh*N_*D_;
  const uint16_t* kbh = kb + (size_t)bh*MPAD*D_;
  const uint16_t* vbh = vt + (size_t)bh*D_*MPAD;

  int nq = n0 + llow; if (nq > N_-1) nq = N_-1;   // clamp ragged last tile
  v16bf aq0 = make_frag(qbh + (size_t)nq*D_ +      kboff);
  v16bf aq1 = make_frag(qbh + (size_t)nq*D_ + 32 + kboff);

  for (int mt = 0; mt < MPAD/16; ++mt) {
    int mrow = mt*16 + llow;
    v8f acc = {};
    acc = wmma_bf16(aq0, make_frag(kbh + (size_t)mrow*D_ +      kboff), acc);
    acc = wmma_bf16(aq1, make_frag(kbh + (size_t)mrow*D_ + 32 + kboff), acc);
    #pragma unroll
    for (int j = 0; j < 8; ++j) {
      int r = j + kboff;
      int c = mt*16 + llow;
      s_sc[r][c] = (c < P_+N_) ? acc[j] * 0.125f : -1e30f;
    }
  }
  __syncthreads();

  if (lane < 16) {
    float mx = -1e30f;
    for (int m = 0; m < P_+N_; ++m) mx = fmaxf(mx, s_sc[lane][m]);
    float sum = 0.f;
    for (int m = 0; m < P_+N_; ++m) sum += __expf(s_sc[lane][m] - mx);
    float inv = 1.0f / sum;
    for (int m = 0; m < MPAD; ++m)
      s_pr[lane][m] = (m < P_+N_) ? f2bf(__expf(s_sc[lane][m] - mx) * inv)
                                  : (uint16_t)0;
  }
  __syncthreads();

  for (int dt = 0; dt < 4; ++dt) {
    v8f acc = {};
    const uint16_t* vrow = vbh + (size_t)(dt*16 + llow)*MPAD;
    for (int mk = 0; mk < MPAD; mk += 32) {
      v16bf ap = make_frag(&s_pr[llow][mk + kboff]);
      v16bf bv = make_frag(vrow + mk + kboff);
      acc = wmma_bf16(ap, bv, acc);
    }
    #pragma unroll
    for (int j = 0; j < 8; ++j) {
      int n = n0 + j + kboff;
      if (n < N_)
        ao[((size_t)(b*N_ + n))*C_ + h*D_ + dt*16 + llow] = f2bf(acc[j]);
    }
  }
}

// ---------- output projection GEMM + bias, f32 out ----------
__global__ void __launch_bounds__(128)
k_gemm_proj(const uint16_t* __restrict__ A,   // attn_out bf16 [MROWS x C]
            const uint16_t* __restrict__ Bt,  // proj_w bf16 [C x C]
            const float* __restrict__ bias,
            float* __restrict__ out) {
  __shared__ __align__(16) uint16_t a_s[2][64][32];
  __shared__ __align__(16) uint16_t b_s[2][128][32];
  const int tid   = threadIdx.x;
  const int wave  = tid >> 5;
  const int wm    = wave & 1, wn = wave >> 1;
  const int lane  = tid & 31;
  const int llow  = lane & 15;
  const int kboff = (lane >> 4) << 3;
  const int m0 = blockIdx.x * 64;
  const int n0 = blockIdx.y * 128;

  v8f acc[8];
  #pragma unroll
  for (int i = 0; i < 8; ++i) acc[i] = (v8f){};

  gemm_mainloop(A + (size_t)m0 * C_, Bt + (size_t)n0 * C_,
                a_s, b_s, tid, wm, wn, llow, kboff, acc);

  #pragma unroll
  for (int r = 0; r < 2; ++r) {
    #pragma unroll
    for (int t = 0; t < 4; ++t) {
      int col = n0 + wn*64 + t*16 + llow;
      float bs = bias[col];
      #pragma unroll
      for (int j = 0; j < 8; ++j) {
        int row = m0 + wm*32 + r*16 + j + kboff;
        out[(size_t)row*C_ + col] = acc[r*4 + t][j] + bs;
      }
    }
  }
}

// ---------- launcher ----------
extern "C" void kernel_launch(void* const* d_in, const int* in_sizes, int n_in,
                              void* d_out, int out_size, void* d_ws, size_t ws_size,
                              hipStream_t stream) {
  const float* x      = (const float*)d_in[0];
  const float* prompt = (const float*)d_in[1];
  const float* qkv_w  = (const float*)d_in[2];
  const float* proj_w = (const float*)d_in[3];
  const float* proj_b = (const float*)d_in[4];
  float* out = (float*)d_out;

  uint16_t* ws = (uint16_t*)d_ws;
  size_t off = 0;
  uint16_t* xbf = ws + off; off += (size_t)MROWS * C_;        // x bf16
  uint16_t* wq  = ws + off; off += (size_t)F_ * C_;           // qkv_w bf16
  uint16_t* wp  = ws + off; off += (size_t)C_ * C_;           // proj_w bf16
  uint16_t* qb  = ws + off; off += (size_t)B_*NH * N_   * D_; // q
  uint16_t* kb  = ws + off; off += (size_t)B_*NH * MPAD * D_; // k (padded)
  uint16_t* vt  = ws + off; off += (size_t)B_*NH * D_ * MPAD; // v^T (padded)
  uint16_t* ao  = ws + off; off += (size_t)MROWS * C_;        // attn out bf16

  k_cvt<<<1024, 256, 0, stream>>>(x,      xbf, MROWS * C_);
  k_cvt<<< 512, 256, 0, stream>>>(qkv_w,  wq,  F_ * C_);
  k_cvt<<< 256, 256, 0, stream>>>(proj_w, wp,  C_ * C_);
  k_prefix<<<B_ * NH, D_, 0, stream>>>(prompt, kb, vt);

  k_gemm_qkv<<<dim3(MROWS/64, F_/128), 128, 0, stream>>>(xbf, wq, qb, kb, vt);
  k_attn<<<dim3((N_+15)/16, NH, B_), 32, 0, stream>>>(qb, kb, vt, ao);
  k_gemm_proj<<<dim3(MROWS/64, C_/128), 128, 0, stream>>>(ao, wp, proj_b, out);
}
